// IntegratedNCA_21689584845379
// MI455X (gfx1250) — matrix operand order repaired and checked by reading, not compiled, not run on hardware
//
#include <hip/hip_runtime.h>
#include <hip/hip_bf16.h>

typedef __attribute__((ext_vector_type(16))) _Float16 v16h;
typedef __attribute__((ext_vector_type(8)))  _Float16 v8h;
typedef __attribute__((ext_vector_type(8)))  float    v8f;

#define NB   8
#define CH   16
#define HH   96
#define WW   96
#define HID  128
#define WDIM 128
#define KC   176     // 16 + 32 + 128 comb channels
#define PLANE (HH*WW)          // 9216
#define IMG   (CH*PLANE)       // per-batch elements
#define NTOT  (NB*IMG)         // 1,179,648 floats

// ---------------------------------------------------------------------------
// Pack W1 (128x176) and W2 (16x128) from f32 row-major into the CDNA5 WMMA
// A-fragment lane layout, with this file's contiguous K order (not ISA table order):
//   lane<16 : M=lane,    half h -> K=h
//   lane>=16: M=lane-16, half h -> K=h+16
// One fragment = 32 lanes x 16 halves, stored contiguously so the step kernel
// loads it with a single v16h (2x b128) per lane.
// ---------------------------------------------------------------------------
__global__ void nca_pack_kernel(const float* __restrict__ W1,
                                const float* __restrict__ W2,
                                _Float16* __restrict__ w1p,
                                _Float16* __restrict__ w2p) {
    int t = blockIdx.x * blockDim.x + threadIdx.x;   // one (fragment, lane)
    if (t >= 52 * 32) return;
    int f    = t >> 5;
    int lane = t & 31;
    int lh   = lane >> 4;       // half-wave
    int mrow = lane & 15;
#pragma unroll
    for (int h = 0; h < 16; ++h) {
        int kl = h + (lh ? 16 : 0);   // A-layout local K
        if (f < 48) {            // W1: 8 M-tiles x 6 K-tiles (K padded to 192)
            int mt = f / 6, kt = f % 6;
            int m = 16 * mt + mrow;
            int k = 32 * kt + kl;
            w1p[(f * 32 + lane) * 16 + h] =
                (k < KC) ? (_Float16)W1[m * KC + k] : (_Float16)0.f;
        } else {                 // W2: 1 M-tile x 4 K-tiles
            int kt = f - 48;
            int k  = 32 * kt + kl;
            w2p[(kt * 32 + lane) * 16 + h] = (_Float16)W2[mrow * HID + k];
        }
    }
}

// ---------------- helpers ----------------
__device__ inline float nca_sobelx(const float* xs, int c, int ry, int rx) {
    const float* p = xs + (c * 10 + ry - 1) * 18 + rx;   // row above
    float t = p[1] - p[-1];
    p += 18;                                             // center row
    float m = p[1] - p[-1];
    p += 18;                                             // row below
    float b = p[1] - p[-1];
    return t + 2.f * m + b;
}
__device__ inline float nca_sobely(const float* xs, int c, int ry, int rx) {
    const float* p = xs + (c * 10 + ry - 1) * 18 + rx;
    float t = p[-1] + 2.f * p[0] + p[1];
    p += 36;
    float b = p[-1] + 2.f * p[0] + p[1];
    return b - t;
}
__device__ inline float nca_rnd01(unsigned int s) {
    s ^= s >> 17; s *= 0xed5ad4bbu;
    s ^= s >> 11; s *= 0xac4c1b51u;
    s ^= s >> 15; s *= 0x31848babu;
    s ^= s >> 14;
    return (float)(s >> 8) * (1.0f / 16777216.0f);
}

// ---------------------------------------------------------------------------
// Step kernel: perception + per-pixel MLP via WMMA, stochastic update.
// Block = 256 threads = 8 waves; each wave = one row of 16 pixels.
// Grid = (WW/16, HH/8, NB).
// ---------------------------------------------------------------------------
__global__ __launch_bounds__(256)
void nca_step_kernel(const float* __restrict__ cur,
                     float* __restrict__ raw,
                     const float* __restrict__ wvec,
                     const _Float16* __restrict__ w1p,
                     const _Float16* __restrict__ w2p,
                     const float* __restrict__ b1,
                     int step) {
    __shared__ float   xs[CH * 10 * 18];        // x tile + 1-px halo
    __shared__ float   wsh[WDIM];               // per-batch w vector
    __shared__ _Float16 hbuf[8 * 16 * HID];     // [wave][pixel n][hid m]

    const int tid = threadIdx.x;
    const int b  = blockIdx.z;
    const int x0 = blockIdx.x * 16;
    const int y0 = blockIdx.y * 8;
    const float* xb = cur + (size_t)b * IMG;

    // cooperative tile load (zero pad outside = conv 'SAME' padding)
    for (int i = tid; i < CH * 10 * 18; i += 256) {
        int c  = i / 180, rem = i % 180;
        int ry = rem / 18, rx = rem % 18;
        int gy = y0 + ry - 1, gx = x0 + rx - 1;
        xs[i] = (gy >= 0 && gy < HH && gx >= 0 && gx < WW)
                    ? xb[(c * HH + gy) * WW + gx] : 0.f;
    }
    if (tid < WDIM) wsh[tid] = wvec[b * WDIM + tid];
    __syncthreads();

    const int wv = tid >> 5, lane = tid & 31;
    const int lh = lane >> 4, n = lane & 15;
    const int ry = wv + 1, rx = n + 1;
    const int y = y0 + wv, xpix = x0 + n;

    // ---- build the six 32x16 B-fragments of the comb matrix --------------
    // B order used here (16-bit 32x16, contiguous K, same as A): lane L, half h -> K = h + 16*(L/16), N = L%16
    v16h bf[6];
#pragma unroll
    for (int kt = 0; kt < 6; ++kt) {
#pragma unroll
        for (int h = 0; h < 16; ++h) {
            int k = 32 * kt + h + 16 * lh;
            float v;
            if (k < 16)        v = xs[(k * 10 + ry) * 18 + rx];
            else if (k < 32)   v = nca_sobelx(xs, k - 16, ry, rx);
            else if (k < 48)   v = nca_sobely(xs, k - 32, ry, rx);
            else if (k < KC)   v = wsh[k - 48];
            else               v = 0.f;
            bf[kt][h] = (_Float16)v;
        }
    }

    // ---- GEMM 1: H(128x16) = W1 * comb, bias + relu, stage to LDS --------
    _Float16* hb = hbuf + wv * (16 * HID);
#pragma unroll
    for (int mt = 0; mt < 8; ++mt) {
        v8f acc = {};
#pragma unroll
        for (int kt = 0; kt < 6; ++kt) {
            v16h a = *(const v16h*)(w1p + ((size_t)(mt * 6 + kt) * 32 + lane) * 16);
            acc = __builtin_amdgcn_wmma_f32_16x16x32_f16(
                false, a, false, bf[kt], (short)0, acc, false, false);
        }
        const float* bp = b1 + 16 * mt + 8 * lh;   // rows m = 16*mt + 8*lh + v
        v8h hv;
#pragma unroll
        for (int v = 0; v < 8; ++v) {
            float t = acc[v] + bp[v];
            hv[v] = (_Float16)(t > 0.f ? t : 0.f);
        }
        // hbuf layout [n][m]: consecutive v -> consecutive m -> one b128 store
        *(v8h*)(hb + n * HID + 16 * mt + 8 * lh) = hv;
    }
    asm volatile("s_wait_dscnt 0" ::: "memory");   // own-wave LDS RAW

    // ---- GEMM 2: delta(16x16) = W2 * H -----------------------------------
    v8f acc2 = {};
#pragma unroll
    for (int kt = 0; kt < 4; ++kt) {
        v16h a   = *(const v16h*)(w2p + ((size_t)kt * 32 + lane) * 16);
        // B-fragment of H: K = 32*kt + 16*lh + h, contiguous in [n][m] layout
        v16h hfr = *(const v16h*)(hb + n * HID + 32 * kt + 16 * lh);
        acc2 = __builtin_amdgcn_wmma_f32_16x16x32_f16(
            false, a, false, hfr, (short)0, acc2, false, false);
    }

    // ---- stochastic update ------------------------------------------------
    float alpha = xs[(3 * 10 + ry) * 18 + rx];
    unsigned int seed =
        (unsigned int)((((step * NB + b) * HH + y) * WW + xpix)) * 2654435761u
        + 0x9e3779b9u;
    float updf = (nca_rnd01(seed) < 0.5f && alpha > 0.1f) ? 1.f : 0.f;

    float* rb = raw + (size_t)b * IMG;
#pragma unroll
    for (int v = 0; v < 8; ++v) {
        int c = v + 8 * lh;                         // C/D layout: M = v + 8*lh
        float xold = xs[(c * 10 + ry) * 18 + rx];
        rb[(c * HH + y) * WW + xpix] = xold + acc2[v] * updf;
    }
}

// ---------------------------------------------------------------------------
// Life-mask kernel: life = (max3x3(old alpha)>0.1) & (max3x3(raw alpha)>0.1)
// (in-bounds-only max == 'SAME' pooling with -inf identity)
// ---------------------------------------------------------------------------
__global__ void nca_mask_kernel(const float* __restrict__ cur,
                                const float* __restrict__ raw,
                                float* __restrict__ nxt) {
    int idx = blockIdx.x * blockDim.x + threadIdx.x;
    if (idx >= NB * PLANE) return;
    int b = idx / PLANE, p = idx % PLANE;
    int y = p / WW, x = p % WW;
    const float* ca = cur + ((size_t)b * CH + 3) * PLANE;
    const float* ra = raw + ((size_t)b * CH + 3) * PLANE;
    float mo = -1e30f, mn = -1e30f;
#pragma unroll
    for (int dy = -1; dy <= 1; ++dy)
#pragma unroll
        for (int dx = -1; dx <= 1; ++dx) {
            int yy = y + dy, xx = x + dx;
            if (yy >= 0 && yy < HH && xx >= 0 && xx < WW) {
                int o = yy * WW + xx;
                mo = fmaxf(mo, ca[o]);
                mn = fmaxf(mn, ra[o]);
            }
        }
    float life = (mo > 0.1f && mn > 0.1f) ? 1.f : 0.f;
    const float* rb = raw + (size_t)b * IMG + p;
    float*       nb = nxt + (size_t)b * IMG + p;
#pragma unroll
    for (int c = 0; c < CH; ++c) nb[c * PLANE] = rb[c * PLANE] * life;
}

// ---------------------------------------------------------------------------
extern "C" void kernel_launch(void* const* d_in, const int* in_sizes, int n_in,
                              void* d_out, int out_size, void* d_ws, size_t ws_size,
                              hipStream_t stream) {
    (void)in_sizes; (void)n_in; (void)out_size; (void)ws_size;
    const float* x  = (const float*)d_in[0];
    const float* w  = (const float*)d_in[1];
    // d_in[2] = sobel weights (values are compile-time constants, unused)
    const float* W1 = (const float*)d_in[3];
    const float* b1 = (const float*)d_in[4];
    const float* W2 = (const float*)d_in[5];
    const int steps = 64;

    float* buf0 = (float*)d_ws;
    float* buf1 = buf0 + NTOT;
    float* rawb = buf1 + NTOT;
    _Float16* w1p = (_Float16*)(rawb + NTOT);
    _Float16* w2p = w1p + 48 * 512;

    hipMemcpyAsync(buf0, x, (size_t)NTOT * sizeof(float),
                   hipMemcpyDeviceToDevice, stream);
    nca_pack_kernel<<<(52 * 32 + 255) / 256, 256, 0, stream>>>(W1, W2, w1p, w2p);

    float* curb = buf0;
    float* nxtb = buf1;
    dim3 gstep(WW / 16, HH / 8, NB);
    int gmask = (NB * PLANE + 255) / 256;
    for (int s = 0; s < steps; ++s) {
        nca_step_kernel<<<gstep, 256, 0, stream>>>(curb, rawb, w, w1p, w2p, b1, s);
        float* dst = (s == steps - 1) ? (float*)d_out : nxtb;
        nca_mask_kernel<<<gmask, 256, 0, stream>>>(curb, rawb, dst);
        float* t = curb; curb = nxtb; nxtb = t;
    }
}
